// MultiHeadAttention_32255204393472
// MI455X (gfx1250) — matrix operand order repaired and checked by reading, not compiled, not run on hardware
//
#include <hip/hip_runtime.h>
#include <hip/hip_bf16.h>

#define D_MODEL   1024
#define NUM_HEADS 16
#define HEAD_DIM  64
#define SEQ       2048
#define BATCH     2
#define SCALE     64.0f   // reference passes scale = head_dim (multiplies!)

typedef __attribute__((ext_vector_type(16))) __bf16 v16bf;
typedef __attribute__((ext_vector_type(8)))  __bf16 v8bf;
typedef __attribute__((ext_vector_type(8)))  float  v8f;

static __device__ __forceinline__ v16bf cat8(v8bf lo, v8bf hi) {
  return __builtin_shufflevector(lo, hi, 0, 1, 2, 3, 4, 5, 6, 7,
                                 8, 9, 10, 11, 12, 13, 14, 15);
}

static __device__ __forceinline__ v8f wmma_bf16(v16bf a, v16bf b, v8f c) {
  // D = A(16x32 bf16) x B(32x16 bf16) + C(16x16 f32)
  return __builtin_amdgcn_wmma_f32_16x16x32_bf16(false, a, false, b, (short)0, c,
                                                 false, false);
}

// Async 16B global -> LDS copy (per-lane addresses), tracked by ASYNCcnt.
static __device__ __forceinline__ void async_g2l_b128(unsigned lds_off,
                                                      const void* gptr) {
  unsigned long long ga = (unsigned long long)gptr;
  asm volatile("global_load_async_to_lds_b128 %0, %1, off"
               :: "v"(lds_off), "v"(ga) : "memory");
}
static __device__ __forceinline__ void wait_async(int n) {
  if (n == 0)
    asm volatile("s_wait_asynccnt 0x0" ::: "memory");
  else
    asm volatile("s_wait_asynccnt 0x2" ::: "memory");
}
static __device__ __forceinline__ void wait_ds0() {
  asm volatile("s_wait_dscnt 0x0" ::: "memory");
}

// ---------------------------------------------------------------------------
// Kernel 0: weight pre-pass.  Wt[which][n][k] = bf16(W[k][n]) -- converts the
// three 1024x1024 f32 weights to bf16 ONCE and transposes them so every WMMA
// B-fragment later is a contiguous 32B load.  LDS-tiled 32x32 transpose.
// ---------------------------------------------------------------------------
__global__ __launch_bounds__(256) void wt_prep_kernel(
    const float* __restrict__ Wq, const float* __restrict__ Wk,
    const float* __restrict__ Wv, __bf16* __restrict__ WtAll) {
  const int which = blockIdx.z;
  const float* W = (which == 0) ? Wq : (which == 1) ? Wk : Wv;
  __bf16* O = WtAll + (size_t)which * D_MODEL * D_MODEL;

  __shared__ float tile[32][33];
  const int k0 = blockIdx.x * 32;
  const int n0 = blockIdx.y * 32;
  const int tx = threadIdx.x & 31;
  const int ty = threadIdx.x >> 5;  // 0..7

#pragma unroll
  for (int i = 0; i < 32; i += 8)
    tile[ty + i][tx] = W[(size_t)(k0 + ty + i) * D_MODEL + n0 + tx];
  __syncthreads();
#pragma unroll
  for (int i = 0; i < 32; i += 8)
    O[(size_t)(n0 + ty + i) * D_MODEL + k0 + tx] = (__bf16)tile[tx][ty + i];
}

// ---------------------------------------------------------------------------
// Kernel 1: QKV projection with a double-buffered async-LDS pipeline.
// Block = 256 threads (8 waves), tile = 128 rows x 64 cols (one head).
// While tile t is consumed out of LDS buffer t&1, tile t+1 streams into the
// other buffer via global_load_async_to_lds_b128.  Async loads complete in
// order, so s_wait_asynccnt <=2 retires exactly the older tile's 2 ops/wave.
// Q/K outputs land bf16 in [B,H,S,dh]; V lands TRANSPOSED in [B,H,dh,S].
// ---------------------------------------------------------------------------
__global__ __launch_bounds__(256) void qkv_proj_kernel(
    const float* __restrict__ q, const float* __restrict__ k,
    const float* __restrict__ v, const __bf16* __restrict__ WtAll,
    __bf16* __restrict__ Qh, __bf16* __restrict__ Kh,
    __bf16* __restrict__ Vt) {
  __shared__ __align__(32) __bf16 wtile[2][64 * 64];  // 2 x 8KB, [n][k]

  const int which = blockIdx.z;
  const float* X = (which == 0) ? q : (which == 1) ? k : v;
  const __bf16* Wt = WtAll + (size_t)which * D_MODEL * D_MODEL;
  __bf16* O = (which == 0) ? Qh : (which == 1) ? Kh : Vt;

  const int wave = threadIdx.x >> 5;
  const int lane = threadIdx.x & 31;
  const int m    = lane & 15;
  const int hi   = lane >> 4;

  const int row0 = blockIdx.x * 128 + wave * 16;  // token rows (over B*S)
  const int h    = blockIdx.y;                    // output head
  const int col0 = h * 64;

  // Per-thread slice of the 8KB tile stage: 2 x 16B.
  const int sn0 = threadIdx.x >> 3;            // row 0..31  (it=0) / +32
  const int skb = (threadIdx.x & 7) * 16;      // byte offset in 128B row

  auto stage = [&](int buf, int kkp) {
#pragma unroll
    for (int it = 0; it < 2; ++it) {
      const int n = sn0 + it * 32;
      unsigned lds = (unsigned)(size_t)&wtile[buf][n * 64] + skb;
      const char* g =
          (const char*)(Wt + ((size_t)(col0 + n) << 10) + kkp) + skb;
      async_g2l_b128(lds, g);
    }
  };

  v8f acc[4] = {};
  stage(0, 0);  // prologue: tile 0 in flight

  for (int kk = 0; kk < D_MODEL; kk += 64) {
    const int buf  = (kk >> 6) & 1;
    const bool more = (kk + 64) < D_MODEL;
    if (more) stage(buf ^ 1, kk + 64);  // overlap next tile with compute
    wait_async(more ? 2 : 0);           // retire THIS tile's async writes
    __syncthreads();                    // ...across all 8 waves

#pragma unroll
    for (int k2 = 0; k2 < 64; k2 += 32) {
      // A fragment: 16x32 of X, f32 -> bf16 (v_cvt_pk_bf16_f32).
      v16bf a;
      const float* xrow = X + (size_t)(row0 + m) * D_MODEL + kk + k2 + hi * 16;
#pragma unroll
      for (int j = 0; j < 8; ++j) {
        const int kb = 2 * j;
        a[2 * j]     = (__bf16)xrow[kb];
        a[2 * j + 1] = (__bf16)xrow[kb + 1];
      }
      // Hoist all 4 B fragments (contiguous 32B/lane from LDS), then WMMA.
      v16bf bfr[4];
#pragma unroll
      for (int f = 0; f < 4; ++f)
        bfr[f] = *(const v16bf*)&wtile[buf][(f * 16 + m) * 64 + k2 + hi * 16];
#pragma unroll
      for (int f = 0; f < 4; ++f) acc[f] = wmma_bf16(a, bfr[f], acc[f]);
    }
    wait_ds0();
    __syncthreads();  // reads done before anyone re-stages this buffer
  }

  // C layout: vgpr r, lane -> (M = r + 8*hi, N = m).
#pragma unroll
  for (int f = 0; f < 4; ++f) {
#pragma unroll
    for (int r = 0; r < 8; ++r) {
      const int row = row0 + r + 8 * hi;  // global token index over B*S
      const int bb  = row >> 11;          // / SEQ
      const int ss  = row & (SEQ - 1);
      const int d   = f * 16 + m;
      const __bf16 val = (__bf16)acc[f][r];
      if (which < 2) {
        O[(((size_t)bb * NUM_HEADS + h) * SEQ + ss) * HEAD_DIM + d] = val;
      } else {  // V transposed: [B, H, dh, S]
        O[(((size_t)bb * NUM_HEADS + h) * HEAD_DIM + d) * SEQ + ss] = val;
      }
    }
  }
}

// ---------------------------------------------------------------------------
// Kernel 2: flash-attention (online softmax).  One wave per (b, h, 16-row
// q-tile); never materializes the 512MB f32 score matrix.  All WMMA operand loads
// are contiguous b128s (K row-major, V transposed).  P is re-laid-out from
// the C layout to the A layout via a DOUBLE-BUFFERED per-wave LDS bounce
// (no read-after-reuse stall).
// ---------------------------------------------------------------------------
__global__ __launch_bounds__(128) void flash_attn_kernel(
    const __bf16* __restrict__ Qh, const __bf16* __restrict__ Kh,
    const __bf16* __restrict__ Vt, float* __restrict__ out) {
  __shared__ __align__(32) __bf16 pbuf[4][2][16 * 32];  // per-wave, 2-deep

  const int wave = threadIdx.x >> 5;
  const int lane = threadIdx.x & 31;
  const int m    = lane & 15;
  const int hi   = lane >> 4;

  const int task = blockIdx.x * 4 + wave;  // 4096 tasks
  const int qt   = task & 127;
  const int h    = (task >> 7) & 15;
  const int b    = task >> 11;

  const size_t headoff = ((size_t)b * NUM_HEADS + h) * SEQ * HEAD_DIM;
  const __bf16* Q = Qh + headoff + (size_t)qt * 16 * HEAD_DIM;
  const __bf16* K = Kh + headoff;
  const __bf16* V = Vt + headoff;  // [dh][S]

  // Preload Q as two A fragments (16 x 64, split into K=32 halves).
  v16bf aq[2];
#pragma unroll
  for (int half = 0; half < 2; ++half) {
    const __bf16* qrow = Q + (size_t)m * HEAD_DIM + half * 32;
    aq[half] = *(const v16bf*)(qrow + hi * 16);
  }

  v8f o[4] = {};
  float mrow[8], lrow[8];
#pragma unroll
  for (int r = 0; r < 8; ++r) { mrow[r] = -3.0e38f; lrow[r] = 0.0f; }

  for (int kt = 0; kt < SEQ; kt += 32) {
    if (kt + 32 < SEQ) {  // global_prefetch_b8 on next K and V tiles
      __builtin_prefetch(K + (size_t)(kt + 32 + lane) * HEAD_DIM, 0, 0);
      __builtin_prefetch(V + (size_t)(lane << 1) * SEQ + kt + 32, 0, 0);
    }

    // ---- S = Q @ K^T for 32 key columns (two 16x16 C frags). -------------
    // B operand is K^T: lane holds key-row (kt + nt*16 + m); contiguous 32B.
    v16bf bk[4];
#pragma unroll
    for (int half = 0; half < 2; ++half) {
      bk[half * 2 + 0] = *(const v16bf*)(K + (size_t)(kt + m) * HEAD_DIM +
                                         half * 32 + hi * 16);
      bk[half * 2 + 1] = *(const v16bf*)(K + (size_t)(kt + 16 + m) * HEAD_DIM +
                                         half * 32 + hi * 16);
    }
    v8f s0 = {}, s1 = {};
#pragma unroll
    for (int half = 0; half < 2; ++half) {
      s0 = wmma_bf16(aq[half], bk[half * 2 + 0], s0);
      s1 = wmma_bf16(aq[half], bk[half * 2 + 1], s1);
    }

    // ---- online softmax (f32); row M = r + 8*hi lives on one 16-lane half
#pragma unroll
    for (int r = 0; r < 8; ++r) {
      float a0 = s0[r] * SCALE;
      float a1 = s1[r] * SCALE;
      float mx = fmaxf(a0, a1);
#pragma unroll
      for (int msk = 8; msk >= 1; msk >>= 1)
        mx = fmaxf(mx, __shfl_xor(mx, msk, 32));
      const float mn = fmaxf(mrow[r], mx);
      const float c  = __expf(mrow[r] - mn);
      const float p0 = __expf(a0 - mn);
      const float p1 = __expf(a1 - mn);
      float rs = p0 + p1;
#pragma unroll
      for (int msk = 8; msk >= 1; msk >>= 1) rs += __shfl_xor(rs, msk, 32);
      lrow[r] = lrow[r] * c + rs;
      mrow[r] = mn;
#pragma unroll
      for (int f = 0; f < 4; ++f) o[f][r] *= c;
      s0[r] = p0;
      s1[r] = p1;
    }

    // ---- relayout P (C layout -> A layout), double-buffered LDS bounce ---
    __bf16* pb = &pbuf[wave][(kt >> 5) & 1][0];
#pragma unroll
    for (int r = 0; r < 8; ++r) {
      pb[(r + 8 * hi) * 32 + m]      = (__bf16)s0[r];
      pb[(r + 8 * hi) * 32 + 16 + m] = (__bf16)s1[r];
    }
    wait_ds0();  // wave-local: stores visible before transposed reads
    const __bf16* prow = pb + m * 32;
    const v16bf ap = *(const v16bf*)(prow + hi * 16);

    // ---- O += P @ V : V^T tile is 4 contiguous-b128 B fragments ----------
    v16bf bv[4];
#pragma unroll
    for (int f = 0; f < 4; ++f)
      bv[f] = *(const v16bf*)(V + (size_t)(f * 16 + m) * SEQ + kt + hi * 16);
#pragma unroll
    for (int f = 0; f < 4; ++f) o[f] = wmma_bf16(ap, bv[f], o[f]);
  }

  // ---- normalize and merge heads: out[b, s, h*64 + d] (f32) --------------
  float* orow =
      out + ((size_t)b * SEQ + (size_t)qt * 16) * D_MODEL + h * HEAD_DIM;
#pragma unroll
  for (int r = 0; r < 8; ++r) {
    const float inv = 1.0f / lrow[r];
#pragma unroll
    for (int f = 0; f < 4; ++f)
      orow[(size_t)(r + 8 * hi) * D_MODEL + f * 16 + m] = o[f][r] * inv;
  }
}

// ---------------------------------------------------------------------------
extern "C" void kernel_launch(void* const* d_in, const int* in_sizes, int n_in,
                              void* d_out, int out_size, void* d_ws,
                              size_t ws_size, hipStream_t stream) {
  (void)in_sizes; (void)n_in; (void)out_size; (void)ws_size;
  const float* q  = (const float*)d_in[0];
  const float* k  = (const float*)d_in[1];
  const float* v  = (const float*)d_in[2];
  const float* Wq = (const float*)d_in[3];
  const float* Wk = (const float*)d_in[4];
  const float* Wv = (const float*)d_in[5];

  const size_t per = (size_t)BATCH * NUM_HEADS * SEQ * HEAD_DIM;  // 4M elems
  __bf16* Qh = (__bf16*)d_ws;                          //  8 MB
  __bf16* Kh = Qh + per;                               //  8 MB
  __bf16* Vt = Kh + per;                               //  8 MB (transposed)
  __bf16* Wt = Vt + per;                               //  6 MB (3 x bf16^T)

  // 0) Weight convert+transpose (once): grid (1024/32, 1024/32, 3).
  wt_prep_kernel<<<dim3(32, 32, 3), 256, 0, stream>>>(Wq, Wk, Wv, Wt);
  // 1) QKV projections: grid (BS/128, heads, 3) x 256 threads.
  qkv_proj_kernel<<<dim3(32, 16, 3), 256, 0, stream>>>(q, k, v, Wt, Qh, Kh,
                                                       Vt);
  // 2) Flash attention: 4096 wave-tasks, 4 waves/block.
  flash_attn_kernel<<<dim3(1024), 128, 0, stream>>>(Qh, Kh, Vt,
                                                    (float*)d_out);
}
